// SNNLoss_53678501265464
// MI455X (gfx1250) — matrix operand order repaired and checked by reading, not compiled, not run on hardware
//
#include <hip/hip_runtime.h>
#include <hip/hip_bf16.h>

#define N 8192
#define DIM 128
#define NCLS 32
#define NCHUNK 16
#define NTILE (N / 16)          // 512 column tiles
#define TJ (NTILE / NCHUNK)     // 32 tiles per chunk

typedef _Float16 v16h __attribute__((ext_vector_type(16)));
typedef _Float16 v8h  __attribute__((ext_vector_type(8)));
typedef float    v8f  __attribute__((ext_vector_type(8)));

__device__ __forceinline__ v16h join16(v8h a, v8h b) {
  return __builtin_shufflevector(a, b, 0,1,2,3,4,5,6,7,8,9,10,11,12,13,14,15);
}

// ---------------------------------------------------------------------------
// Kernel 1: row-normalize x and split xn into f16 hi + f16 lo correction.
// One block (128 threads = 4 waves) per row.
// ---------------------------------------------------------------------------
__global__ void __launch_bounds__(128)
snn_normalize_split(const float* __restrict__ x,
                    _Float16* __restrict__ xh,
                    _Float16* __restrict__ xl) {
  __shared__ float red[4];
  __shared__ float s_inv;
  const int row = blockIdx.x;
  const int t = threadIdx.x;                 // 0..127
  float v = x[row * DIM + t];
  float s = v * v;
  #pragma unroll
  for (int off = 16; off >= 1; off >>= 1) s += __shfl_xor(s, off, 32);
  if ((t & 31) == 0) red[t >> 5] = s;
  __syncthreads();
  if (t == 0) {
    float tot = red[0] + red[1] + red[2] + red[3];
    s_inv = 1.0f / fmaxf(sqrtf(tot), 1e-8f);
  }
  __syncthreads();
  const float xn = v * s_inv;
  const _Float16 hi = (_Float16)xn;
  const _Float16 lo = (_Float16)(xn - (float)hi);
  xh[row * DIM + t] = hi;
  xl[row * DIM + t] = lo;
}

// ---------------------------------------------------------------------------
// Kernel 2: fused WMMA GEMM tile + exp + masked pos/neg row sums.
// One wave (32 threads) per (16-row block bi, column chunk).
// S tile = A(16x128) * B(128x16) via split-f16: hi*hi + hi*lo + lo*hi.
// ---------------------------------------------------------------------------
__global__ void __launch_bounds__(32)
snn_tile(const _Float16* __restrict__ xh, const _Float16* __restrict__ xl,
         const int* __restrict__ targets, const float* __restrict__ temperature,
         float* __restrict__ posp, float* __restrict__ negp) {
  const int lane = threadIdx.x;   // 0..31
  const int half = lane >> 4;     // K-subset / row-subset selector
  const int l15  = lane & 15;
  const int bi    = blockIdx.x;   // row tile index (0..511)
  const int chunk = blockIdx.y;   // column chunk (0..15)
  const int rbase = bi * 16;
  const float invT = 1.0f / temperature[0];

  // --- A fragments for full K=128 (ISA 16-bit A layout: lanes 0-15 hold
  // K {k0..k0+7} in v0-3 and {k0+16..k0+23} in v4-7; lanes 16-31 the +8 sets).
  v16h Ahi[4], Alo[4];
  const int arow = rbase + l15;
  #pragma unroll
  for (int kc = 0; kc < 4; ++kc) {
    const _Float16* p = xh + arow * DIM + kc * 32 + half * 8;
    Ahi[kc] = join16(*(const v8h*)p, *(const v8h*)(p + 16));
    const _Float16* q = xl + arow * DIM + kc * 32 + half * 8;
    Alo[kc] = join16(*(const v8h*)q, *(const v8h*)(q + 16));
  }

  // Row classes for the D-matrix layout: VGPR v <-> global row rbase+v+8*half.
  int trow[8];
  #pragma unroll
  for (int v = 0; v < 8; ++v) trow[v] = targets[rbase + v + 8 * half];

  float accP[8], accN[8];
  #pragma unroll
  for (int v = 0; v < 8; ++v) { accP[v] = 0.0f; accN[v] = 0.0f; }

  const int jt0 = chunk * TJ;
  for (int jt = jt0; jt < jt0 + TJ; ++jt) {
    const int cbase = jt * 16;
    const int col = cbase + l15;   // this lane's column (B lane = N index)

    v8f Dacc = {};
    #pragma unroll
    for (int kc = 0; kc < 4; ++kc) {
      // B layout: lane n, same K order as A: {kc*32+half*8..+7} then {+16..+23}.
      const _Float16* pb = xh + col * DIM + kc * 32 + half * 8;
      v16h Bhi = join16(*(const v8h*)pb, *(const v8h*)(pb + 16));
      const _Float16* qb = xl + col * DIM + kc * 32 + half * 8;
      v16h Blo = join16(*(const v8h*)qb, *(const v8h*)(qb + 16));
      Dacc = __builtin_amdgcn_wmma_f32_16x16x32_f16(false, Ahi[kc], false, Bhi,
                                                    (short)0, Dacc, false, false);
      Dacc = __builtin_amdgcn_wmma_f32_16x16x32_f16(false, Ahi[kc], false, Blo,
                                                    (short)0, Dacc, false, false);
      Dacc = __builtin_amdgcn_wmma_f32_16x16x32_f16(false, Alo[kc], false, Bhi,
                                                    (short)0, Dacc, false, false);
    }

    const int tcol = targets[col];
    #pragma unroll
    for (int v = 0; v < 8; ++v) {
      const float s = Dacc[v] * invT;
      const float e = __expf(s);
      const int rowg = rbase + v + 8 * half;
      const bool same = (trow[v] == tcol);
      // Skipping the diagonal == reference's "pos_sum - exp(S_ii)".
      accP[v] += (same && (rowg != col)) ? e : 0.0f;
      accN[v] += same ? 0.0f : e;
    }
  }

  // Reduce over the 16 columns held by each half-wave; store per-chunk
  // partials (each (chunk,row) slot written exactly once -> deterministic).
  #pragma unroll
  for (int v = 0; v < 8; ++v) {
    float p = accP[v], n = accN[v];
    #pragma unroll
    for (int off = 8; off >= 1; off >>= 1) {
      p += __shfl_xor(p, off, 16);
      n += __shfl_xor(n, off, 16);
    }
    if (l15 == 0) {
      const int row = rbase + v + 8 * half;
      posp[chunk * N + row] = p;
      negp[chunk * N + row] = n;
    }
  }
}

// ---------------------------------------------------------------------------
// Kernel 3: finalize — class histogram, per-row loss, deterministic
// per-class reduction, inverse-frequency weighting -> scalar loss.
// ---------------------------------------------------------------------------
__global__ void __launch_bounds__(64)
snn_finalize(const float* __restrict__ posp, const float* __restrict__ negp,
             const int* __restrict__ targets, float* __restrict__ out) {
  __shared__ int   counts[NCLS];
  __shared__ float lsum[64 * NCLS];
  __shared__ float lcnt[64 * NCLS];
  const int t = threadIdx.x;                 // 0..63
  if (t < NCLS) counts[t] = 0;
  for (int c = 0; c < NCLS; ++c) { lsum[t * NCLS + c] = 0.0f; lcnt[t * NCLS + c] = 0.0f; }
  __syncthreads();
  for (int i = t; i < N; i += 64) atomicAdd(&counts[targets[i]], 1);  // integer: deterministic
  __syncthreads();
  for (int i = t; i < N; i += 64) {
    float p = 0.0f, n = 0.0f;
    #pragma unroll
    for (int ch = 0; ch < NCHUNK; ++ch) { p += posp[ch * N + i]; n += negp[ch * N + i]; }
    const int c = targets[i];
    const int cnt = counts[c];
    const bool valid = (cnt >= 2) && ((N - cnt) >= 1);
    const float ps = valid ? p : 1.0f;
    const float ns = valid ? n : 1.0f;
    const float loss = -logf(ps / (ps + ns));
    lsum[t * NCLS + c] += valid ? loss : 0.0f;   // private slot: no race
    lcnt[t * NCLS + c] += valid ? 1.0f : 0.0f;
  }
  __syncthreads();
  for (int s = 32; s >= 1; s >>= 1) {
    if (t < s) {
      for (int c = 0; c < NCLS; ++c) {
        lsum[t * NCLS + c] += lsum[(t + s) * NCLS + c];
        lcnt[t * NCLS + c] += lcnt[(t + s) * NCLS + c];
      }
    }
    __syncthreads();
  }
  if (t == 0) {
    float sumInv = 0.0f;
    for (int c = 0; c < NCLS; ++c)
      sumInv += (counts[c] > 0) ? 1.0f / fmaxf((float)counts[c], 1.0f) : 0.0f;
    float acc = 0.0f, npres = 0.0f;
    for (int c = 0; c < NCLS; ++c) {
      const float cc = lcnt[c];
      const bool present = cc > 0.0f;
      const float mean = lsum[c] / fmaxf(cc, 1.0f);
      const float inv = (counts[c] > 0) ? 1.0f / fmaxf((float)counts[c], 1.0f) : 0.0f;
      const float w = (sumInv > 0.0f) ? inv / sumInv : 0.0f;
      acc += present ? mean * w : 0.0f;
      npres += present ? 1.0f : 0.0f;
    }
    out[0] = acc / fmaxf(npres, 1.0f);
  }
}

// ---------------------------------------------------------------------------
extern "C" void kernel_launch(void* const* d_in, const int* in_sizes, int n_in,
                              void* d_out, int out_size, void* d_ws, size_t ws_size,
                              hipStream_t stream) {
  (void)in_sizes; (void)n_in; (void)out_size; (void)ws_size;
  const float* x           = (const float*)d_in[0];
  const int*   targets     = (const int*)d_in[1];
  const float* temperature = (const float*)d_in[2];
  float* out = (float*)d_out;

  // Workspace layout: xh (2 MB) | xl (2 MB) | posp (512 KB) | negp (512 KB)
  _Float16* xh = (_Float16*)d_ws;
  _Float16* xl = xh + (size_t)N * DIM;
  float* posp = (float*)(xl + (size_t)N * DIM);
  float* negp = posp + (size_t)NCHUNK * N;

  snn_normalize_split<<<dim3(N), dim3(DIM), 0, stream>>>(x, xh, xl);
  snn_tile<<<dim3(N / 16, NCHUNK), dim3(32), 0, stream>>>(xh, xl, targets,
                                                          temperature, posp, negp);
  snn_finalize<<<dim3(1), dim3(64), 0, stream>>>(posp, negp, targets, out);
}
